// SparseBoostingMoE_8100308320514
// MI455X (gfx1250) — matrix operand order repaired and checked by reading, not compiled, not run on hardware
//
#include <hip/hip_runtime.h>

// ---------------------------------------------------------------------------
// SparseBoostingMoE on gfx1250 (MI455X): grouped-GEMM MoE with bf16 WMMA and
// async global->LDS staging (GLOBAL_LOAD_ASYNC_TO_LDS_B128 / ASYNCcnt).
// D=1024, H=2048, E=8, T=4096 tokens, TOP_K=2, ALPHA=0.5, LN eps=1e-5.
// ---------------------------------------------------------------------------

typedef __bf16  v16bf __attribute__((ext_vector_type(16)));
typedef float   v8f   __attribute__((ext_vector_type(8)));

struct FragBytes { uint4 lo, hi; };   // 32 bytes == one v16bf fragment

__device__ __forceinline__ unsigned short f2bf(float f) {
    unsigned u = __builtin_bit_cast(unsigned, f);
    unsigned r = u + 0x7FFFu + ((u >> 16) & 1u);   // round-to-nearest-even
    return (unsigned short)(r >> 16);
}

static constexpr int Dh   = 1024;   // hidden
static constexpr int Hh   = 2048;   // expert hidden
static constexpr int Eh   = 8;      // experts
static constexpr int Th   = 4096;   // tokens (2*2048)
static constexpr int PMAX = 4352;   // max padded rows (4096 + 8*16, rounded up)

// ---------------------------------------------------------------------------
// 0) init counters + perm tables
// ---------------------------------------------------------------------------
__global__ void moe_init(int* cnt0, int* cnt1, int* pos0, int* pos1,
                         int* perm0, int* perm1) {
    int i = blockIdx.x * blockDim.x + threadIdx.x;
    if (i < Eh) { cnt0[i] = 0; cnt1[i] = 0; pos0[i] = 0; pos1[i] = 0; }
    if (i < PMAX) { perm0[i] = -1; perm1[i] = -1; }
}

// ---------------------------------------------------------------------------
// 1) gating: one wave32 per token. logits -> softmax -> top2 -> counts
// ---------------------------------------------------------------------------
__global__ __launch_bounds__(256)
void moe_gate(const float* __restrict__ x, const float* __restrict__ gW,
              const float* __restrict__ gb, float* __restrict__ topv,
              int* __restrict__ topi, int* __restrict__ cnt0,
              int* __restrict__ cnt1) {
    const int wave = threadIdx.x >> 5, lane = threadIdx.x & 31;
    const int t = blockIdx.x * 8 + wave;
    if (t >= Th) return;
    const float* xr = x + (size_t)t * Dh;
    float acc[Eh] = {};
    for (int d = lane; d < Dh; d += 32) {
        float xv = xr[d];
        const float* g = gW + d * Eh;
        #pragma unroll
        for (int e = 0; e < Eh; ++e) acc[e] += xv * g[e];
    }
    #pragma unroll
    for (int e = 0; e < Eh; ++e)
        #pragma unroll
        for (int o = 16; o > 0; o >>= 1) acc[e] += __shfl_xor(acc[e], o, 32);
    if (lane == 0) {
        float w[Eh], mx = -3.0e38f;
        #pragma unroll
        for (int e = 0; e < Eh; ++e) { w[e] = acc[e] + gb[e]; mx = fmaxf(mx, w[e]); }
        float s = 0.f;
        #pragma unroll
        for (int e = 0; e < Eh; ++e) { w[e] = expf(w[e] - mx); s += w[e]; }
        float inv = 1.f / s;
        #pragma unroll
        for (int e = 0; e < Eh; ++e) w[e] *= inv;
        int i0 = 0;
        #pragma unroll
        for (int e = 1; e < Eh; ++e) if (w[e] > w[i0]) i0 = e;
        int i1 = (i0 == 0) ? 1 : 0;
        #pragma unroll
        for (int e = 0; e < Eh; ++e) if (e != i0 && w[e] > w[i1]) i1 = e;
        topv[2 * t] = w[i0]; topv[2 * t + 1] = w[i1];
        topi[2 * t] = i0;    topi[2 * t + 1] = i1;
        atomicAdd(&cnt0[i0], 1);
        atomicAdd(&cnt1[i1], 1);
    }
}

// ---------------------------------------------------------------------------
// 2) tiny per-expert offset prefix sum (16-row aligned segments)
// ---------------------------------------------------------------------------
__global__ void moe_offsets(const int* cnt0, const int* cnt1,
                            int* offs0, int* offs1) {
    if (threadIdx.x == 0 && blockIdx.x == 0) {
        int o = 0;
        for (int e = 0; e < Eh; ++e) { offs0[e] = o; o += (cnt0[e] + 15) & ~15; }
        o = 0;
        for (int e = 0; e < Eh; ++e) { offs1[e] = o; o += (cnt1[e] + 15) & ~15; }
    }
}

// ---------------------------------------------------------------------------
// 3) pack weights fp32 -> bf16 in WMMA B-fragment-ready layout, coalesced.
//    Block = one (e, kt, 64-col chunk): load 32x64 fp32 chunk with 256B
//    coalesced runs -> bf16 in LDS (stride 66 to spread banks) -> each
//    64-thread group emits one 32x16 fragment tile as contiguous 16B stores.
//    Tile layout: dst = (((e*NT + nt)*KT + kt)*32 + lane)*16 + j with
//    k = kt*32 + (lane>>4)*8 + (j>>3)*16 + (j&7) ; n = nt*16 + (lane&15)
// ---------------------------------------------------------------------------
__global__ __launch_bounds__(256)
void moe_pack_w(const float* __restrict__ W, unsigned short* __restrict__ dst,
                int K, int N) {
    const int KT = K >> 5, NT = N >> 4;
    const int nChunks = N >> 6;
    int bid = blockIdx.x;
    const int kt = bid % KT; bid /= KT;
    const int nc = bid % nChunks;
    const int e  = bid / nChunks;
    const int nBase = nc * 64;

    __shared__ unsigned short sb[32 * 66];            // [k_local][n_local] pad 2
    const float* src = W + ((size_t)e * K + (size_t)kt * 32) * N + nBase;
    #pragma unroll
    for (int r = 0; r < 8; ++r) {                     // 2048 floats, coalesced
        int idx = r * 256 + threadIdx.x;
        int kl = idx >> 6, nl = idx & 63;
        sb[kl * 66 + nl] = f2bf(src[(size_t)kl * N + nl]);
    }
    __syncthreads();

    const int tileLocal = threadIdx.x >> 6;           // n-tile 0..3 in chunk
    const int sub  = threadIdx.x & 63;                // 64 threads per tile
    const int lane = sub >> 1;                        // fragment lane 0..31
    const int jBase = (sub & 1) * 8;                  // low/high 8 elements
    const int nt = (nBase >> 4) + tileLocal;
    const int nl = tileLocal * 16 + (lane & 15);
    const int kBase = (lane >> 4) * 8 + jBase * 2;
    unsigned short tmp[8];
    #pragma unroll
    for (int j = 0; j < 8; ++j)
        tmp[j] = sb[(kBase + j) * 66 + nl];
    unsigned short* out = dst +
        (((size_t)e * NT + nt) * (size_t)KT + kt) * 512 + lane * 16 + jBase;
    *(uint4*)out = *(const uint4*)tmp;
}

// ---------------------------------------------------------------------------
// 4) gather tokens into per-expert contiguous bf16 rows (+ optional boost)
// ---------------------------------------------------------------------------
__global__ __launch_bounds__(256)
void moe_gather(const float* __restrict__ x, const float* __restrict__ eoPrev,
                float alpha, const int* __restrict__ topi, int which,
                const int* __restrict__ offs, int* __restrict__ pos,
                int* __restrict__ perm, unsigned short* __restrict__ Xe) {
    const int t = blockIdx.x;
    __shared__ int srow;
    if (threadIdx.x == 0) {
        int e = topi[2 * t + which];
        int p = atomicAdd(&pos[e], 1);
        int row = offs[e] + p;
        perm[row] = t;
        srow = row;
    }
    __syncthreads();
    const int row = srow;
    const float* xr = x + (size_t)t * Dh;
    const float* er = eoPrev ? eoPrev + (size_t)t * Dh : nullptr;
    unsigned short* dst = Xe + (size_t)row * Dh;
    for (int d = threadIdx.x; d < Dh; d += 256) {
        float v = xr[d];
        if (er) v += alpha * er[d];
        dst[d] = f2bf(v);
    }
}

// ---------------------------------------------------------------------------
// 5) grouped GEMM on WMMA bf16.
//    Block = 8 waves: 256 rows x 64 cols. Wave: 2 m-tiles x 4 n-tiles
//    (16 WMMAs per k-pair, 1 ds_load_b128 per WMMA).
//    B tiles staged with GLOBAL_LOAD_ASYNC_TO_LDS_B128 (ASYNCcnt), double
//    buffered, 2 k-slices per buffer -> DMA fully overlaps WMMA compute;
//    wave only waits s_wait_asynccnt right before the barrier.
//    FIRST:  He = relu(A @ W1[e] + b1[e])  (bf16 out, row-major)
//    !FIRST: eo[tok] = A @ W2[e] + b2[e]   (fp32, scattered via perm)
// ---------------------------------------------------------------------------
template <bool FIRST>
__global__ __launch_bounds__(256)
void moe_gemm(const unsigned short* __restrict__ A,
              const unsigned short* __restrict__ Bfrag,
              const float* __restrict__ bias,
              unsigned short* __restrict__ outBf,
              float* __restrict__ outScat,
              const int* __restrict__ cnt, const int* __restrict__ offs,
              const int* __restrict__ perm, int K, int N) {
    const int e = blockIdx.z;
    const int rows = cnt[e];
    const int prows = (rows + 15) & ~15;
    const int mBlock = blockIdx.y * 256;
    if (mBlock >= prows) return;                      // uniform early-out
    const int off  = offs[e];
    const int tid  = threadIdx.x;
    const int wave = tid >> 5, lane = tid & 31;
    const int half = lane >> 4, mlane = lane & 15;
    const int nt0  = blockIdx.x * 4;                  // 4 n-tiles per block
    const int KT   = K >> 5;                          // 32-wide k tiles
    const int KS   = K >> 6;                          // staged k-pairs
    const unsigned short* Be =
        Bfrag + (size_t)e * (size_t)(N >> 4) * (size_t)KT * 512;

    // [buf][ (nt*2 + kk) * 512 ] fragment tiles, 16KB total
    __shared__ __align__(16) unsigned short lds[2][8 * 512];

    const int  mWave = mBlock + wave * 32;            // 2 m-tiles per wave
    const bool mv0 = mWave < prows;
    const bool mv1 = mWave + 16 < prows;
    const int  mu0 = mv0 ? mWave : prows - 16;        // clamp (discarded)
    const int  mu1 = mv1 ? mWave + 16 : prows - 16;
    // A fragment base per ISA 16-bit A layout: row = lane&15,
    // K groups {0..7,16..23} (+8 for lanes 16..31)
    const unsigned short* Arow0 =
        A + (size_t)(off + mu0 + mlane) * (size_t)K + half * 8;
    const unsigned short* Arow1 =
        A + (size_t)(off + mu1 + mlane) * (size_t)K + half * 8;

    // staging: 256 threads move 8 tiles * 1KB per k-pair; 2 x b128 per thread
    const int stTile = tid >> 6;                      // n tile 0..3
    const int stSub  = tid & 63;                      // 64 thr * 16B = 1 tile
    const unsigned short* Bst =
        Be + (size_t)(nt0 + stTile) * (size_t)KT * 512 + stSub * 8;
    // generic shared ptr truncated to 32 bits == wave-relative LDS address
    unsigned ldsDst[2][2];
    #pragma unroll
    for (int b = 0; b < 2; ++b)
        #pragma unroll
        for (int kk = 0; kk < 2; ++kk)
            ldsDst[b][kk] = (unsigned)(unsigned long long)
                &lds[b][(stTile * 2 + kk) * 512 + stSub * 8];

    auto stage = [&](int b, int ks) {
        #pragma unroll
        for (int kk = 0; kk < 2; ++kk) {
            unsigned long long g = (unsigned long long)
                (Bst + (size_t)(ks * 2 + kk) * 512);
            asm volatile("global_load_async_to_lds_b128 %0, %1, off"
                         :: "v"(ldsDst[b][kk]), "v"(g) : "memory");
        }
    };

    v8f acc0[4] = {}, acc1[4] = {};

    stage(0, 0);
    asm volatile("s_wait_asynccnt 0x0" ::: "memory");
    __syncthreads();

    for (int ks = 0; ks < KS; ++ks) {
        const int buf = ks & 1;
        if (ks + 1 < KS) stage(buf ^ 1, ks + 1);      // DMA overlaps compute
        #pragma unroll
        for (int kk = 0; kk < 2; ++kk) {
            const int kt = ks * 2 + kk;
            FragBytes fa0, fa1;
            fa0.lo = *(const uint4*)(Arow0 + kt * 32);
            fa0.hi = *(const uint4*)(Arow0 + kt * 32 + 16);
            fa1.lo = *(const uint4*)(Arow1 + kt * 32);
            fa1.hi = *(const uint4*)(Arow1 + kt * 32 + 16);
            v16bf a0 = __builtin_bit_cast(v16bf, fa0);
            v16bf a1 = __builtin_bit_cast(v16bf, fa1);
            const unsigned short* lb = &lds[buf][kk * 512 + lane * 16];
            #pragma unroll
            for (int tt = 0; tt < 4; ++tt) {
                FragBytes fb;
                fb.lo = *(const uint4*)(lb + tt * 1024);
                fb.hi = *(const uint4*)(lb + tt * 1024 + 8);
                v16bf b = __builtin_bit_cast(v16bf, fb);
                acc0[tt] = __builtin_amdgcn_wmma_f32_16x16x32_bf16(
                    false, a0, false, b, (short)0, acc0[tt], false, false);
                acc1[tt] = __builtin_amdgcn_wmma_f32_16x16x32_bf16(
                    false, a1, false, b, (short)0, acc1[tt], false, false);
            }
        }
        asm volatile("s_wait_asynccnt 0x0" ::: "memory");
        __syncthreads();
    }

    auto epilogue = [&](bool mValid, int mW, v8f (&acc)[4]) {
        if (!mValid) return;
        const int nCol  = nt0 * 16 + mlane;           // + tt*16 per tile
        const int rbase = off + mW + half * 8;        // C VGPR r -> row rbase+r
        if (FIRST) {
            #pragma unroll
            for (int tt = 0; tt < 4; ++tt) {
                const float bv = bias[e * N + nCol + tt * 16];
                #pragma unroll
                for (int r = 0; r < 8; ++r) {
                    float v = acc[tt][r] + bv;
                    v = v > 0.f ? v : 0.f;            // relu
                    outBf[(size_t)(rbase + r) * (size_t)N + nCol + tt * 16] =
                        f2bf(v);
                }
            }
        } else {
            int tok[8];
            #pragma unroll
            for (int r = 0; r < 8; ++r) tok[r] = perm[rbase + r];
            #pragma unroll
            for (int tt = 0; tt < 4; ++tt) {
                const float bv = bias[e * N + nCol + tt * 16];
                #pragma unroll
                for (int r = 0; r < 8; ++r)
                    if (tok[r] >= 0)
                        outScat[(size_t)tok[r] * (size_t)N + nCol + tt * 16] =
                            acc[tt][r] + bv;
            }
        }
    };
    epilogue(mv0, mWave, acc0);
    epilogue(mv1, mWave + 16, acc1);
}

// ---------------------------------------------------------------------------
// 6) fused epilogue: y = x + v0*eo0 + v1*eo1 ; out = LayerNorm(y)
// ---------------------------------------------------------------------------
__global__ __launch_bounds__(256)
void moe_final(const float* __restrict__ x, const float* __restrict__ eo0,
               const float* __restrict__ eo1, const float* __restrict__ topv,
               const float* __restrict__ gamma, const float* __restrict__ beta,
               float* __restrict__ out) {
    const int t = blockIdx.x;
    __shared__ float ybuf[Dh];
    __shared__ float red[256];
    const float v0 = topv[2 * t], v1 = topv[2 * t + 1];
    const size_t b = (size_t)t * Dh;
    float s = 0.f;
    for (int d = threadIdx.x; d < Dh; d += 256) {
        float y = x[b + d] + v0 * eo0[b + d] + v1 * eo1[b + d];
        ybuf[d] = y;
        s += y;
    }
    red[threadIdx.x] = s; __syncthreads();
    for (int o = 128; o > 0; o >>= 1) {
        if (threadIdx.x < o) red[threadIdx.x] += red[threadIdx.x + o];
        __syncthreads();
    }
    const float mu = red[0] / Dh;
    __syncthreads();
    float sq = 0.f;
    for (int d = threadIdx.x; d < Dh; d += 256) {
        float c = ybuf[d] - mu;
        sq += c * c;
    }
    red[threadIdx.x] = sq; __syncthreads();
    for (int o = 128; o > 0; o >>= 1) {
        if (threadIdx.x < o) red[threadIdx.x] += red[threadIdx.x + o];
        __syncthreads();
    }
    const float rstd = rsqrtf(red[0] / Dh + 1e-5f);
    for (int d = threadIdx.x; d < Dh; d += 256)
        out[b + d] = (ybuf[d] - mu) * rstd * gamma[d] + beta[d];
}

// ---------------------------------------------------------------------------
extern "C" void kernel_launch(void* const* d_in, const int* in_sizes, int n_in,
                              void* d_out, int out_size, void* d_ws,
                              size_t ws_size, hipStream_t stream) {
    (void)in_sizes; (void)n_in; (void)out_size; (void)ws_size;
    const float* x      = (const float*)d_in[0];
    const float* gate_W = (const float*)d_in[1];
    const float* gate_b = (const float*)d_in[2];
    const float* W1     = (const float*)d_in[3];
    const float* b1     = (const float*)d_in[4];
    const float* W2     = (const float*)d_in[5];
    const float* b2     = (const float*)d_in[6];
    const float* ln_g   = (const float*)d_in[7];
    const float* ln_b   = (const float*)d_in[8];
    float* out = (float*)d_out;

    char* ws = (char*)d_ws;
    size_t o = 0;
    auto take = [&](size_t bytes) -> char* {
        char* p = ws + o;
        o = (o + bytes + 255) & ~(size_t)255;
        return p;
    };
    unsigned short* W1bf = (unsigned short*)take((size_t)Eh * Dh * Hh * 2);
    unsigned short* W2bf = (unsigned short*)take((size_t)Eh * Hh * Dh * 2);
    unsigned short* Xe   = (unsigned short*)take((size_t)PMAX * Dh * 2);
    unsigned short* He   = (unsigned short*)take((size_t)PMAX * Hh * 2);
    float* eo0  = (float*)take((size_t)Th * Dh * 4);
    float* eo1  = (float*)take((size_t)Th * Dh * 4);
    float* topv = (float*)take((size_t)Th * 2 * 4);
    int*   topi = (int*)take((size_t)Th * 2 * 4);
    int* cnt0 = (int*)take(Eh * 4); int* cnt1 = (int*)take(Eh * 4);
    int* pos0 = (int*)take(Eh * 4); int* pos1 = (int*)take(Eh * 4);
    int* offs0 = (int*)take(Eh * 4); int* offs1 = (int*)take(Eh * 4);
    int* perm0 = (int*)take(PMAX * 4); int* perm1 = (int*)take(PMAX * 4);

    // 0) init
    moe_init<<<(PMAX + 255) / 256, 256, 0, stream>>>(cnt0, cnt1, pos0, pos1,
                                                     perm0, perm1);
    // 1) gating + counting
    moe_gate<<<Th / 8, 256, 0, stream>>>(x, gate_W, gate_b, topv, topi,
                                         cnt0, cnt1);
    // 2) segment offsets
    moe_offsets<<<1, 32, 0, stream>>>(cnt0, cnt1, offs0, offs1);
    // 3) pack weights to bf16 fragment layout (coalesced through LDS)
    moe_pack_w<<<Eh * (Dh / 32) * (Hh / 64), 256, 0, stream>>>(W1, W1bf,
                                                               Dh, Hh);
    moe_pack_w<<<Eh * (Hh / 32) * (Dh / 64), 256, 0, stream>>>(W2, W2bf,
                                                               Hh, Dh);

    const dim3 blk(256);
    const dim3 g1(Hh / 64, 16, Eh);   // GEMM1: N=2048, 256 rows per block
    const dim3 g2(Dh / 64, 16, Eh);   // GEMM2: N=1024

    // ---- pass 0 ----
    moe_gather<<<Th, 256, 0, stream>>>(x, nullptr, 0.f, topi, 0,
                                       offs0, pos0, perm0, Xe);
    moe_gemm<true><<<g1, blk, 0, stream>>>(Xe, W1bf, b1, He, nullptr,
                                           cnt0, offs0, perm0, Dh, Hh);
    moe_gemm<false><<<g2, blk, 0, stream>>>(He, W2bf, b2, nullptr, eo0,
                                            cnt0, offs0, perm0, Hh, Dh);
    // ---- pass 1 (boost = x + 0.5*eo0) ----
    moe_gather<<<Th, 256, 0, stream>>>(x, eo0, 0.5f, topi, 1,
                                       offs1, pos1, perm1, Xe);
    moe_gemm<true><<<g1, blk, 0, stream>>>(Xe, W1bf, b1, He, nullptr,
                                           cnt1, offs1, perm1, Dh, Hh);
    moe_gemm<false><<<g2, blk, 0, stream>>>(He, W2bf, b2, nullptr, eo1,
                                            cnt1, offs1, perm1, Hh, Dh);
    // ---- fused residual + gate-weighted sum + LayerNorm ----
    moe_final<<<Th, 256, 0, stream>>>(x, eo0, eo1, topv, ln_g, ln_b, out);
}
